// CausalMultiHeadSelfAttention_89206470738273
// MI455X (gfx1250) — matrix operand order repaired and checked by reading, not compiled, not run on hardware
//
#include <hip/hip_runtime.h>

// ---------- types ----------
typedef _Float16 h16;
typedef __attribute__((ext_vector_type(16))) _Float16 v16h;
typedef __attribute__((ext_vector_type(8)))  _Float16 v8h;
typedef __attribute__((ext_vector_type(8)))  float    v8f;
typedef __attribute__((ext_vector_type(4)))  int      i32x4;

// address-space-qualified pointee types for the async-LDS builtin
// (attribute bound via typedef so it is not dropped as a decl attribute)
typedef __attribute__((address_space(1))) i32x4 as1_i32x4;
typedef __attribute__((address_space(3))) i32x4 as3_i32x4;

#define D_MODEL 1024
#define N_HEAD 16
#define HEAD_DIM 64
#define BATCH 2
#define SEQ 2048

#if __has_builtin(__builtin_amdgcn_global_load_async_to_lds_b128)
#define USE_ASYNC_LDS 1
#else
#define USE_ASYNC_LDS 0
#endif

// ---------- WMMA helpers (CDNA5 16x16x32 f16 -> f32) ----------
__device__ __forceinline__ v8f wmma_f16(v16h a, v16h b, v8f c) {
    return __builtin_amdgcn_wmma_f32_16x16x32_f16(
        /*neg_a=*/false, a, /*neg_b=*/false, b,
        /*c_mod=*/(short)0, c, /*reuse_a=*/false, /*reuse_b=*/false);
}

// A-fragment: 16x32 f16, element i of v16h holds K = (i/8)*16 + grp*8 + (i%8)
// for row M = lane&15 (grp = lane>>4). From row-major [M x K] storage this is
// two contiguous 16-byte loads per lane.
__device__ __forceinline__ v16h frag_a(const h16* base, int ld) {
    const int lane = threadIdx.x & 31;
    const h16* p = base + (size_t)(lane & 15) * ld + ((lane >> 4) << 3);
    v8h lo = *reinterpret_cast<const v8h*>(p);
    v8h hi = *reinterpret_cast<const v8h*>(p + 16);
    v16h r;
#pragma unroll
    for (int i = 0; i < 8; ++i) { r[i] = lo[i]; r[i + 8] = hi[i]; }
    return r;
}

// B-fragment: 32x16 f16, element i of v16h holds K = (i/8)*16 + grp*8 + (i%8) for
// column N = lane&15. From storage where K is contiguous per N
// (index = n*ld + k) this is two contiguous 16-byte loads per lane.
__device__ __forceinline__ v16h frag_b(const h16* base, int ld) {
    const int lane = threadIdx.x & 31;
    const h16* p = base + (size_t)(lane & 15) * ld + ((lane >> 4) << 3);
    v8h lo = *reinterpret_cast<const v8h*>(p);
    v8h hi = *reinterpret_cast<const v8h*>(p + 16);
    v16h r;
#pragma unroll
    for (int i = 0; i < 8; ++i) { r[i] = lo[i]; r[i + 8] = hi[i]; }
    return r;
}

// ---------- async-copy helpers ----------
__device__ __forceinline__ void async_copy16(const h16* src, h16* dst) {
#if USE_ASYNC_LDS
    void* s = (void*)src;   // strip const, then retype with address spaces
    void* d = (void*)dst;
    __builtin_amdgcn_global_load_async_to_lds_b128((as1_i32x4*)s, (as3_i32x4*)d, 0, 0);
#else
    *reinterpret_cast<v8h*>(dst) = *reinterpret_cast<const v8h*>(src);
#endif
}

__device__ __forceinline__ void wait_async_le2() {
#if __has_builtin(__builtin_amdgcn_s_wait_asynccnt)
    __builtin_amdgcn_s_wait_asynccnt(2);
#else
    asm volatile("s_wait_asynccnt 0x2" ::: "memory");
#endif
}
__device__ __forceinline__ void wait_async_le0() {
#if __has_builtin(__builtin_amdgcn_s_wait_asynccnt)
    __builtin_amdgcn_s_wait_asynccnt(0);
#else
    asm volatile("s_wait_asynccnt 0x0" ::: "memory");
#endif
}

// Stage a 64(N) x 32(K) f16 tile of Bt (row-major [N][K], ld=Kd) into LDS
// (ld=32). 128 threads x 2 chunks of 16 B each = 4 KB.
__device__ __forceinline__ void stage_b(const h16* __restrict__ Bt, h16* buf,
                                        int n0, int k0, int Kd) {
    const int tid = threadIdx.x;
#pragma unroll
    for (int i = 0; i < 2; ++i) {
        const int c = tid + i * 128;          // chunk id 0..255
        const int row = c >> 2;               // 0..63
        const int part = c & 3;               // 16-byte part within row
        const h16* src = Bt + (size_t)(n0 + row) * Kd + k0 + part * 8;
        h16* dst = buf + row * 32 + part * 8;
        async_copy16(src, dst);
    }
}

// ---------- conversion kernels ----------
__global__ void k_f32_to_f16(const float* __restrict__ in, h16* __restrict__ out, int n) {
    int i = blockIdx.x * blockDim.x + threadIdx.x;
    if (i < n) out[i] = (h16)in[i];
}

// W[k][n] (row-major in,out) -> Wt[n][k] f16 (K contiguous per output column)
__global__ void k_w_transpose(const float* __restrict__ W, h16* __restrict__ Wt, int D) {
    int i = blockIdx.x * blockDim.x + threadIdx.x;
    if (i < D * D) {
        int n = i / D, k = i - n * D;
        Wt[(size_t)n * D + k] = (h16)W[(size_t)k * D + n];
    }
}

// ---------- WMMA GEMM: C[M,N] = A[M,K] * Bt[N,K]^T ----------
// Block = 128 threads = 4 waves; block tile 128(M) x 64(N); each wave 32x64.
// B tile (64x32) staged to LDS once per block via async DMA, double-buffered.
// MODE 0: store f16 row-major   (Q, K projections)
// MODE 1: store f32 row-major   (final output projection)
// MODE 2: store f16 scattered as Vt[b][h][dh][s]  (V projection)
template <int MODE>
__global__ __launch_bounds__(128) void k_gemm(const h16* __restrict__ A,
                                              const h16* __restrict__ Bt,
                                              void* __restrict__ outp,
                                              int M, int N, int Kd) {
    __shared__ h16 bbuf[2][64 * 32];          // 2 x 4 KB double buffer

    const int wave = threadIdx.x >> 5;
    const int lane = threadIdx.x & 31;
    const int m0 = (blockIdx.y * 4 + wave) * 32;
    const int n0 = blockIdx.x * 64;

    v8f acc0[4], acc1[4];
#pragma unroll
    for (int j = 0; j < 4; ++j)
#pragma unroll
        for (int r = 0; r < 8; ++r) { acc0[j][r] = 0.0f; acc1[j][r] = 0.0f; }

    const h16* Ab = A + (size_t)m0 * Kd;
    const int nK = Kd >> 5;

    stage_b(Bt, &bbuf[0][0], n0, 0, Kd);
    for (int kt = 0; kt < nK; ++kt) {
        const int cur = kt & 1;
        if (kt + 1 < nK) {
            stage_b(Bt, &bbuf[cur ^ 1][0], n0, (kt + 1) << 5, Kd);
            wait_async_le2();                  // this wave's buf[cur] loads done
        } else {
            wait_async_le0();
        }
        __syncthreads();                       // all waves' DMA into buf[cur] visible

        const int k0 = kt << 5;
        v16h a0 = frag_a(Ab + k0, Kd);
        v16h a1 = frag_a(Ab + (size_t)16 * Kd + k0, Kd);
#pragma unroll
        for (int j = 0; j < 4; ++j) {
            v16h b = frag_b(&bbuf[cur][j * 16 * 32], 32);
            acc0[j] = wmma_f16(a0, b, acc0[j]);
            acc1[j] = wmma_f16(a1, b, acc1[j]);
        }
        __syncthreads();                       // done reading buf[cur] before overwrite
    }

    const int grp = lane >> 4, nc = lane & 15;
#pragma unroll
    for (int g = 0; g < 2; ++g) {
#pragma unroll
        for (int j = 0; j < 4; ++j) {
#pragma unroll
            for (int r = 0; r < 8; ++r) {
                int m = m0 + g * 16 + r + grp * 8;
                int n = n0 + j * 16 + nc;
                float v = (g == 0) ? acc0[j][r] : acc1[j][r];
                if (MODE == 0) {
                    ((h16*)outp)[(size_t)m * N + n] = (h16)v;
                } else if (MODE == 1) {
                    ((float*)outp)[(size_t)m * N + n] = v;
                } else {
                    int b = m >> 11, s = m & (SEQ - 1);
                    int hh = n >> 6, dh = n & (HEAD_DIM - 1);
                    ((h16*)outp)[((size_t)((b * N_HEAD + hh) * HEAD_DIM + dh)) * SEQ + s] = (h16)v;
                }
            }
        }
    }
}

// ---------- flash attention (causal), one wave per 16-query tile ----------
__global__ __launch_bounds__(256) void k_attn(const h16* __restrict__ Q,
                                              const h16* __restrict__ K,
                                              const h16* __restrict__ Vt,
                                              h16* __restrict__ ctx) {
    __shared__ h16 ldsP[8][16 * 32];   // per-wave P staging (f16), 8 KB total

    const int wave = threadIdx.x >> 5;
    const int lane = threadIdx.x & 31;
    const int wg = blockIdx.x * 8 + wave;          // [0, B*H*S/16)
    const int qt = wg & (SEQ / 16 - 1);            // 128 query tiles
    const int h  = (wg >> 7) & (N_HEAD - 1);
    const int bb = wg >> 11;
    const int q0 = qt * 16;

    const h16* Qb = Q  + (size_t)(bb * SEQ + q0) * D_MODEL + h * HEAD_DIM;
    const h16* Kb = K  + (size_t)(bb * SEQ) * D_MODEL + h * HEAD_DIM;
    const h16* Vb = Vt + (size_t)((bb * N_HEAD + h) * HEAD_DIM) * SEQ;

    // Q fragments (16 queries x 64 dims = two K-chunks), kept in registers
    v16h aQ0 = frag_a(Qb, D_MODEL);
    v16h aQ1 = frag_a(Qb + 32, D_MODEL);

    float mrun[8], lrun[8];
    v8f o[4];
#pragma unroll
    for (int r = 0; r < 8; ++r) { mrun[r] = -INFINITY; lrun[r] = 0.0f; }
#pragma unroll
    for (int j = 0; j < 4; ++j)
#pragma unroll
        for (int r = 0; r < 8; ++r) o[j][r] = 0.0f;

    const int grp = lane >> 4, kc = lane & 15;
    const float scale = 0.125f;                    // 1/sqrt(64)
    const int nkt = (q0 + 47) >> 5;                // 32-key tiles, covers diagonal

    h16* myP = &ldsP[wave][0];

    for (int t = 0; t < nkt; ++t) {
        const int k0 = t * 32;
        const bool last = (t == nkt - 1);

        // scores for 16 queries x 32 keys (two 16-key WMMA tiles)
        v8f s0, s1;
#pragma unroll
        for (int r = 0; r < 8; ++r) { s0[r] = 0.0f; s1[r] = 0.0f; }
        {
            v16h b00 = frag_b(Kb + (size_t)k0 * D_MODEL, D_MODEL);
            v16h b01 = frag_b(Kb + (size_t)k0 * D_MODEL + 32, D_MODEL);
            s0 = wmma_f16(aQ0, b00, s0);
            s0 = wmma_f16(aQ1, b01, s0);
            v16h b10 = frag_b(Kb + (size_t)(k0 + 16) * D_MODEL, D_MODEL);
            v16h b11 = frag_b(Kb + (size_t)(k0 + 16) * D_MODEL + 32, D_MODEL);
            s1 = wmma_f16(aQ0, b10, s1);
            s1 = wmma_f16(aQ1, b11, s1);
        }

        // online softmax over the 32 new keys, per query row
#pragma unroll
        for (int r = 0; r < 8; ++r) {
            const int qr = r + grp * 8;            // query row within tile
            float v0 = s0[r] * scale;
            float v1 = s1[r] * scale;
            if (last) {                            // causal mask (data select, EXEC stays full)
                v0 = (k0 + kc      > q0 + qr) ? -INFINITY : v0;
                v1 = (k0 + 16 + kc > q0 + qr) ? -INFINITY : v1;
            }
            float mx = fmaxf(v0, v1);
#pragma unroll
            for (int msk = 1; msk < 16; msk <<= 1)
                mx = fmaxf(mx, __shfl_xor(mx, msk, 32));
            const float nm = fmaxf(mrun[r], mx);
            const float p0 = __expf(v0 - nm);
            const float p1 = __expf(v1 - nm);
            float sum = p0 + p1;
#pragma unroll
            for (int msk = 1; msk < 16; msk <<= 1)
                sum += __shfl_xor(sum, msk, 32);
            const float alpha = __expf(mrun[r] - nm);
            lrun[r] = lrun[r] * alpha + sum;
            mrun[r] = nm;
#pragma unroll
            for (int j = 0; j < 4; ++j) o[j][r] *= alpha;
            // stage P (f16) row-major 16x32 in LDS
            myP[qr * 32 + kc]      = (h16)p0;
            myP[qr * 32 + 16 + kc] = (h16)p1;
        }

        // P (A-frag from LDS) x V (B-frag from Vt: keys contiguous per dim)
        v16h aP = frag_a(myP, 32);
#pragma unroll
        for (int j = 0; j < 4; ++j) {
            v16h bv = frag_b(Vb + (size_t)(j * 16) * SEQ + k0, SEQ);
            o[j] = wmma_f16(aP, bv, o[j]);
        }
    }

    // normalize and store ctx (f16 row-major [B*S, D], heads interleaved)
#pragma unroll
    for (int j = 0; j < 4; ++j) {
#pragma unroll
        for (int r = 0; r < 8; ++r) {
            const int qr = r + grp * 8;
            ctx[(size_t)(bb * SEQ + q0 + qr) * D_MODEL + h * HEAD_DIM + j * 16 + kc] =
                (h16)(o[j][r] / lrun[r]);
        }
    }
}

// ---------- host launcher ----------
extern "C" void kernel_launch(void* const* d_in, const int* in_sizes, int n_in,
                              void* d_out, int out_size, void* d_ws, size_t ws_size,
                              hipStream_t stream) {
    (void)in_sizes; (void)n_in; (void)out_size; (void)ws_size;

    const float* x  = (const float*)d_in[0];
    const float* Wq = (const float*)d_in[1];
    const float* Wk = (const float*)d_in[2];
    const float* Wv = (const float*)d_in[3];
    const float* Wo = (const float*)d_in[4];

    const int M = BATCH * SEQ;                // 4096
    const size_t MD = (size_t)M * D_MODEL;    // 4,194,304 elems
    const size_t DD = (size_t)D_MODEL * D_MODEL;

    char* ws = (char*)d_ws;
    size_t off = 0;
    h16* Xh  = (h16*)(ws + off); off += MD * 2;   // 8 MB
    h16* Wqt = (h16*)(ws + off); off += DD * 2;   // 2 MB
    h16* Wkt = (h16*)(ws + off); off += DD * 2;
    h16* Wvt = (h16*)(ws + off); off += DD * 2;
    h16* Wot = (h16*)(ws + off); off += DD * 2;
    h16* Qh  = (h16*)(ws + off); off += MD * 2;   // 8 MB
    h16* Kh  = (h16*)(ws + off); off += MD * 2;
    h16* Vt  = (h16*)(ws + off); off += MD * 2;
    h16* Ctx = (h16*)(ws + off); off += MD * 2;   // total ~48 MB

    // 1) precision conversion + weight transposes
    k_f32_to_f16<<<(int)((MD + 255) / 256), 256, 0, stream>>>(x, Xh, (int)MD);
    k_w_transpose<<<(int)((DD + 255) / 256), 256, 0, stream>>>(Wq, Wqt, D_MODEL);
    k_w_transpose<<<(int)((DD + 255) / 256), 256, 0, stream>>>(Wk, Wkt, D_MODEL);
    k_w_transpose<<<(int)((DD + 255) / 256), 256, 0, stream>>>(Wv, Wvt, D_MODEL);
    k_w_transpose<<<(int)((DD + 255) / 256), 256, 0, stream>>>(Wo, Wot, D_MODEL);

    // 2) QKV projections (WMMA GEMMs). Block = 128x64 tile, 4 waves.
    dim3 gg(D_MODEL / 64, M / 128);
    k_gemm<0><<<gg, 128, 0, stream>>>(Xh, Wqt, (void*)Qh, M, D_MODEL, D_MODEL);
    k_gemm<0><<<gg, 128, 0, stream>>>(Xh, Wkt, (void*)Kh, M, D_MODEL, D_MODEL);
    k_gemm<2><<<gg, 128, 0, stream>>>(Xh, Wvt, (void*)Vt, M, D_MODEL, D_MODEL);

    // 3) causal flash attention: B*H*(S/16) = 4096 waves, 8 waves/block
    k_attn<<<(BATCH * N_HEAD * (SEQ / 16)) / 8, 256, 0, stream>>>(Qh, Kh, Vt, Ctx);

    // 4) output projection -> fp32 result
    k_gemm<1><<<gg, 128, 0, stream>>>(Ctx, Wot, d_out, M, D_MODEL, D_MODEL);
}
